// AttModel_vis_grid_40235253629203
// MI455X (gfx1250) — matrix operand order repaired and checked by reading, not compiled, not run on hardware
//
#include <hip/hip_runtime.h>

// ---------------------------------------------------------------------------
// Model constants (match the JAX reference)
// ---------------------------------------------------------------------------
#define B_   64
#define LV_  196
#define LQ_  20
#define SS_  216        // LV + LQ
#define H_   768
#define NH_  12
#define DH_  64
#define NB_  6
#define FF_  3072
#define QB_  14         // ceil(216/16) query blocks
#define KPAD_ 320       // padded glove K (300 -> 320)
#define NTOK_ (B_ * SS_)          // 13824
#define SCALE_ 0.125f             // 1/sqrt(64)

// ---------------------------------------------------------------------------
// WMMA types / helpers (gfx1250, wave32)
// ---------------------------------------------------------------------------
typedef __bf16          v16bf __attribute__((ext_vector_type(16)));
typedef unsigned short  v16us __attribute__((ext_vector_type(16)));
typedef unsigned short  v8us  __attribute__((ext_vector_type(8)));
typedef float           v8f   __attribute__((ext_vector_type(8)));

// fp32 -> bf16, round-to-nearest (ties away): add bias, take high half.
__device__ __forceinline__ unsigned short f2bf(float f) {
  return (unsigned short)((__builtin_bit_cast(unsigned, f) + 0x8000u) >> 16);
}

// Pack two fp32 into packed bf16 pair with a single v_perm_b32.
__device__ __forceinline__ unsigned pack2(float a, float b) {
  unsigned ua = __builtin_bit_cast(unsigned, a) + 0x8000u;
  unsigned ub = __builtin_bit_cast(unsigned, b) + 0x8000u;
  // D = { ub[31:16], ua[31:16] }
  return __builtin_amdgcn_perm(ub, ua, 0x07060302u);
}

// Two aligned 16B LDS loads -> one 16-half fragment (ds_load_b128 x2)
__device__ __forceinline__ v16us ld_frag(const unsigned short* p0,
                                         const unsigned short* p1) {
  v8us lo = *(const v8us*)p0;
  v8us hi = *(const v8us*)p1;
  return __builtin_shufflevector(lo, hi, 0, 1, 2, 3, 4, 5, 6, 7,
                                 8, 9, 10, 11, 12, 13, 14, 15);
}

__device__ __forceinline__ v8f bf_wmma(v16us a, v16us b, v8f c) {
  return __builtin_amdgcn_wmma_f32_16x16x32_bf16(
      false, __builtin_bit_cast(v16bf, a),
      false, __builtin_bit_cast(v16bf, b),
      (short)0, c, false, false);
}

// ---------------------------------------------------------------------------
// GEMM: C = act(A[MxK] * Bw[KxN] + bias), bf16 WMMA, fp32 accumulate.
// REQUIRES: M%64==0, N%64==0, K%32==0 (guaranteed by the launcher).
// fp32 operands converted to bf16 while staging into double-buffered LDS;
// B staged TRANSPOSED (Bst[n][k]) so all fragments are contiguous b128 loads.
// Staging is load-phase / store-phase split so all 8 global_load_b128 of a
// tile are in flight before any conversion waits on them.
// Output row remap: orow = (row/rpb)*obs + roff + row%rpb.
// ---------------------------------------------------------------------------
__global__ __launch_bounds__(128)
void gemm_bf16_wmma(const float* __restrict__ A, const float* __restrict__ Bw,
                    const float* __restrict__ bias, float* __restrict__ C,
                    int M, int N, int K, int relu,
                    int rpb, int obs, int roff) {
  __shared__ alignas(16) unsigned short As[2][2048];   // [m][k]
  __shared__ alignas(16) unsigned short Bst[2][2048];  // [n][k] transposed

  const int tid  = threadIdx.x;
  const int lane = tid & 31;
  const int w    = tid >> 5;
  const int wm   = w >> 1, wn = w & 1;     // 2x2 wave grid, 32x32 per wave
  const int hi   = lane >> 4;
  const int ln   = lane & 15;
  const int bM   = blockIdx.y * 64, bN = blockIdx.x * 64;

  // Stage one 64x32 A tile + 32x64 B tile: gather all loads, then store.
  auto stage = [&](int buf, int k0) {
    float4 av[4];                        // A: 4 float4 per thread
#pragma unroll
    for (int t = 0; t < 4; ++t) {
      int idx = tid + t * 128;
      int r = idx >> 3, c4 = idx & 7;
      av[t] = *reinterpret_cast<const float4*>(
          A + (size_t)(bM + r) * K + k0 + c4 * 4);
    }
    float4 bva[2], bvb[2];               // B: 2 row-pairs per thread
#pragma unroll
    for (int t = 0; t < 2; ++t) {
      int idx = tid + t * 128;
      int rp = idx >> 4, c4 = idx & 15;
      const float* base = Bw + (size_t)(k0 + 2 * rp) * N + bN + c4 * 4;
      bva[t] = *reinterpret_cast<const float4*>(base);
      bvb[t] = *reinterpret_cast<const float4*>(base + N);
    }
#pragma unroll
    for (int t = 0; t < 4; ++t) {
      int idx = tid + t * 128;
      int r = idx >> 3, c4 = idx & 7;
      *(uint2*)&As[buf][r * 32 + c4 * 4] =
          make_uint2(pack2(av[t].x, av[t].y), pack2(av[t].z, av[t].w));
    }
#pragma unroll
    for (int t = 0; t < 2; ++t) {
      int idx = tid + t * 128;
      int rp = idx >> 4, c4 = idx & 15;
      unsigned short* dst = &Bst[buf][(c4 * 4) * 32 + 2 * rp];
      *(unsigned*)(dst + 0 * 32) = pack2(bva[t].x, bvb[t].x);
      *(unsigned*)(dst + 1 * 32) = pack2(bva[t].y, bvb[t].y);
      *(unsigned*)(dst + 2 * 32) = pack2(bva[t].z, bvb[t].z);
      *(unsigned*)(dst + 3 * 32) = pack2(bva[t].w, bvb[t].w);
    }
  };

  v8f acc[2][2] = {{{}, {}}, {{}, {}}};

  stage(0, 0);
  __syncthreads();

  int cur = 0;
  for (int k0 = 0; k0 < K; k0 += 32) {
    const int kn = k0 + 32;
    if (kn < K) {                       // stage next tile into other buffer
      stage(cur ^ 1, kn);
      if (kn + 32 < K) {                // speculative prefetch 2 tiles ahead
        int pa = bM + (tid >> 1);
        __builtin_prefetch(&A[(size_t)pa * K + kn + 32], 0, 1);
        int pb = kn + 32 + (tid >> 2);
        __builtin_prefetch(&Bw[(size_t)pb * N + bN + (tid & 3) * 16], 0, 1);
      }
    }
    const unsigned short* arow0 = &As[cur][(wm * 32 + ln) * 32];
    const unsigned short* arow1 = &As[cur][(wm * 32 + 16 + ln) * 32];
    v16us a0 = ld_frag(arow0 + hi * 8, arow0 + 16 + hi * 8);
    v16us a1 = ld_frag(arow1 + hi * 8, arow1 + 16 + hi * 8);
    const unsigned short* brow0 = &Bst[cur][(wn * 32 + ln) * 32];
    const unsigned short* brow1 = &Bst[cur][(wn * 32 + 16 + ln) * 32];
    v16us b0 = ld_frag(brow0 + hi * 8, brow0 + 16 + hi * 8);
    v16us b1 = ld_frag(brow1 + hi * 8, brow1 + 16 + hi * 8);

    acc[0][0] = bf_wmma(a0, b0, acc[0][0]);
    acc[0][1] = bf_wmma(a0, b1, acc[0][1]);
    acc[1][0] = bf_wmma(a1, b0, acc[1][0]);
    acc[1][1] = bf_wmma(a1, b1, acc[1][1]);
    __syncthreads();
    cur ^= 1;
  }

  // Epilogue: bias + relu + incremental row remap (one div per fragment row)
  const int coln = bN + wn * 32 + ln;
  float b0v = bias ? bias[coln] : 0.f;
  float b1v = bias ? bias[coln + 16] : 0.f;
#pragma unroll
  for (int mi = 0; mi < 2; ++mi) {
    int base = bM + wm * 32 + mi * 16 + hi * 8;
    int q = base / rpb, rem = base % rpb;
#pragma unroll
    for (int r = 0; r < 8; ++r) {
      int orow = q * obs + roff + rem;
      float* crow = C + (size_t)orow * N + coln;
      float v0 = acc[mi][0][r] + b0v;
      float v1 = acc[mi][1][r] + b1v;
      if (relu) { v0 = fmaxf(v0, 0.f); v1 = fmaxf(v1, 0.f); }
      crow[0]  = v0;
      crow[16] = v1;
      if (++rem == rpb) { rem = 0; ++q; }
    }
  }
}

// ---------------------------------------------------------------------------
// Encoder attention: one wave per (b, h, 16-query block). WMMA for both
// Q@K^T and P@V; masked softmax in LDS. Q/K/V buffers are padded by 8 zeroed
// rows so all staging loads are unguarded b128s. O may alias Q.
// ---------------------------------------------------------------------------
__global__ __launch_bounds__(32)
void attn_wmma(const float* __restrict__ Q, const float* __restrict__ Kk,
               const float* __restrict__ V, const float* __restrict__ mask,
               float* __restrict__ O) {
  __shared__ alignas(16) unsigned short qs[16 * 64];     // [q][d]
  __shared__ alignas(16) unsigned short kvbuf[64 * 32];  // K: [kk16][d64]; V: [d64][kk32]
  __shared__ alignas(16) float          sc[16 * 224];
  __shared__ alignas(16) unsigned short pbuf[16 * 224];

  const int id = blockIdx.x;
  const int qb = id % QB_;
  const int h  = (id / QB_) % NH_;
  const int b  = id / (QB_ * NH_);
  const int lane = threadIdx.x;
  const int hi = lane >> 4, ln = lane & 15;

  // Stage a 16-row x 64-col fp32 block as bf16 [row][d]: load-all-then-store.
  auto stage16 = [&](const float* src, int row0, unsigned short* dst) {
    float4 v[8];
#pragma unroll
    for (int t = 0; t < 8; ++t) {
      int idx = lane + t * 32;
      int r = idx >> 4, c4 = idx & 15;
      v[t] = *reinterpret_cast<const float4*>(
          src + ((size_t)(b * SS_ + row0 + r)) * H_ + h * DH_ + c4 * 4);
    }
#pragma unroll
    for (int t = 0; t < 8; ++t) {
      int idx = lane + t * 32;
      int r = idx >> 4, c4 = idx & 15;
      *(uint2*)&dst[r * 64 + c4 * 4] =
          make_uint2(pack2(v[t].x, v[t].y), pack2(v[t].z, v[t].w));
    }
  };

  stage16(Q, qb * 16, qs);
  __syncthreads();

  const unsigned short* qrow = &qs[ln * 64];
  v16us aq0 = ld_frag(qrow + hi * 8, qrow + 16 + hi * 8);
  v16us aq1 = ld_frag(qrow + 32 + hi * 8, qrow + 48 + hi * 8);

  // Scores: loop 16-key tiles; kvbuf holds K tile [kk][d]
  for (int kt = 0; kt < QB_; ++kt) {
    __syncthreads();
    stage16(Kk, kt * 16, kvbuf);
    __syncthreads();
    const unsigned short* krow = &kvbuf[ln * 64];
    v16us b0 = ld_frag(krow + hi * 8, krow + 16 + hi * 8);
    v16us b1 = ld_frag(krow + 32 + hi * 8, krow + 48 + hi * 8);
    v8f acc = {};
    acc = bf_wmma(aq0, b0, acc);
    acc = bf_wmma(aq1, b1, acc);
#pragma unroll
    for (int r = 0; r < 8; ++r) {
      int m = hi * 8 + r;
      int kkg = kt * 16 + ln;
      int qg = qb * 16 + m;
      float s = acc[r] * SCALE_;
      if (kkg >= SS_ || qg >= SS_) {
        s = -1e9f;
      } else {
        float mv = mask[((size_t)(b * SS_ + qg)) * SS_ + kkg];
        if (!(mv > 0.f)) s = -1e9f;
      }
      sc[m * 224 + kt * 16 + ln] = s;
    }
  }
  __syncthreads();

  // Row softmax over 216 valid keys (lanes 0..15 each own a row)
  if (lane < 16) {
    int m = lane;
    float mx = -1e30f;
    for (int j = 0; j < SS_; ++j) mx = fmaxf(mx, sc[m * 224 + j]);
    float sum = 0.f;
    for (int j = 0; j < SS_; ++j) {
      float e = __expf(sc[m * 224 + j] - mx);
      sc[m * 224 + j] = e;
      sum += e;
    }
    float inv = 1.f / sum;
    for (int j = 0; j < SS_; ++j) pbuf[m * 224 + j] = f2bf(sc[m * 224 + j] * inv);
    for (int j = SS_; j < 224; ++j) pbuf[m * 224 + j] = 0;
  }
  __syncthreads();

  // O = P @ V : 7 chunks of 32 keys; V staged TRANSPOSED ([d][kk]).
  // Staging done in two 4-pair groups: loads batched, then converted.
  v8f oacc[4] = {{}, {}, {}, {}};
  for (int ck = 0; ck < 7; ++ck) {
    __syncthreads();
#pragma unroll
    for (int g = 0; g < 2; ++g) {
      float4 va[4], vb[4];
#pragma unroll
      for (int t = 0; t < 4; ++t) {
        int idx = lane + (g * 4 + t) * 32;
        int rp = idx >> 4, c4 = idx & 15;     // rp: key-pair, c4: d/4
        const float* vbase =
            V + ((size_t)(b * SS_ + ck * 32 + 2 * rp)) * H_ + h * DH_ + c4 * 4;
        va[t] = *reinterpret_cast<const float4*>(vbase);
        vb[t] = *reinterpret_cast<const float4*>(vbase + H_);
      }
#pragma unroll
      for (int t = 0; t < 4; ++t) {
        int idx = lane + (g * 4 + t) * 32;
        int rp = idx >> 4, c4 = idx & 15;
        unsigned short* dst = &kvbuf[(c4 * 4) * 32 + 2 * rp];
        *(unsigned*)(dst + 0 * 32) = pack2(va[t].x, vb[t].x);
        *(unsigned*)(dst + 1 * 32) = pack2(va[t].y, vb[t].y);
        *(unsigned*)(dst + 2 * 32) = pack2(va[t].z, vb[t].z);
        *(unsigned*)(dst + 3 * 32) = pack2(va[t].w, vb[t].w);
      }
    }
    __syncthreads();
    const unsigned short* prow = &pbuf[ln * 224 + ck * 32];
    v16us ap = ld_frag(prow + hi * 8, prow + 16 + hi * 8);
#pragma unroll
    for (int nt = 0; nt < 4; ++nt) {
      const unsigned short* vrow = &kvbuf[(nt * 16 + ln) * 32];
      v16us bv = ld_frag(vrow + hi * 8, vrow + 16 + hi * 8);
      oacc[nt] = bf_wmma(ap, bv, oacc[nt]);
    }
  }
#pragma unroll
  for (int nt = 0; nt < 4; ++nt)
#pragma unroll
    for (int r = 0; r < 8; ++r) {
      int m = hi * 8 + r;
      int qg = qb * 16 + m;
      if (qg < SS_)
        O[((size_t)(b * SS_ + qg)) * H_ + h * DH_ + nt * 16 + ln] = oacc[nt][r];
    }
}

// ---------------------------------------------------------------------------
// out = LayerNorm(x + r), H=768, one wave per token (wave32 shuffles)
// ---------------------------------------------------------------------------
__global__ __launch_bounds__(256)
void ln_residual(const float* __restrict__ X, const float* __restrict__ R,
                 float* __restrict__ Out, int ntok) {
  int wid = threadIdx.x >> 5;
  int lane = threadIdx.x & 31;
  int tok = blockIdx.x * 8 + wid;
  if (tok >= ntok) return;
  const float* x = X + (size_t)tok * H_;
  const float* r = R + (size_t)tok * H_;
  float buf[24];
  float s = 0.f;
#pragma unroll
  for (int t = 0; t < 24; ++t) {
    float v = x[lane + t * 32] + r[lane + t * 32];
    buf[t] = v;
    s += v;
  }
#pragma unroll
  for (int o = 16; o > 0; o >>= 1) s += __shfl_xor(s, o, 32);
  float mean = s * (1.f / H_);
  float vs = 0.f;
#pragma unroll
  for (int t = 0; t < 24; ++t) { float d = buf[t] - mean; vs += d * d; }
#pragma unroll
  for (int o = 16; o > 0; o >>= 1) vs += __shfl_xor(vs, o, 32);
  float inv = rsqrtf(vs * (1.f / H_) + 1e-6f);
  float* out = Out + (size_t)tok * H_;
#pragma unroll
  for (int t = 0; t < 24; ++t) out[lane + t * 32] = (buf[t] - mean) * inv;
}

// ---------------------------------------------------------------------------
// Decoder cross attention (1 query token, 216 keys), one wave per (b,h)
// ---------------------------------------------------------------------------
__global__ __launch_bounds__(32)
void dec_cross_attn(const float* __restrict__ Qd, const float* __restrict__ Kc,
                    const float* __restrict__ Vc, float* __restrict__ O) {
  __shared__ float qd[64];
  __shared__ float pr[224];
  const int h = blockIdx.x % NH_;
  const int b = blockIdx.x / NH_;
  const int lane = threadIdx.x;
  qd[lane]      = Qd[(size_t)b * H_ + h * DH_ + lane];
  qd[lane + 32] = Qd[(size_t)b * H_ + h * DH_ + lane + 32];
  __syncthreads();
  float mx = -1e30f;
  for (int j = lane; j < SS_; j += 32) {
    const float* kr = Kc + ((size_t)(b * SS_ + j)) * H_ + h * DH_;
    float dot = 0.f;
#pragma unroll 16
    for (int d = 0; d < 64; ++d) dot += qd[d] * kr[d];
    dot *= SCALE_;
    pr[j] = dot;
    mx = fmaxf(mx, dot);
  }
#pragma unroll
  for (int o = 16; o > 0; o >>= 1) mx = fmaxf(mx, __shfl_xor(mx, o, 32));
  __syncthreads();
  float sum = 0.f;
  for (int j = lane; j < SS_; j += 32) {
    float e = __expf(pr[j] - mx);
    pr[j] = e;
    sum += e;
  }
#pragma unroll
  for (int o = 16; o > 0; o >>= 1) sum += __shfl_xor(sum, o, 32);
  float inv = 1.f / sum;
  __syncthreads();
  float o0 = 0.f, o1 = 0.f;
  for (int j = 0; j < SS_; ++j) {
    const float* vr = Vc + ((size_t)(b * SS_ + j)) * H_ + h * DH_;
    float p = pr[j] * inv;
    o0 += p * vr[lane];
    o1 += p * vr[lane + 32];
  }
  O[(size_t)b * H_ + h * DH_ + lane]      = o0;
  O[(size_t)b * H_ + h * DH_ + lane + 32] = o1;
}

// ---------------------------------------------------------------------------
// Small prep kernels
// ---------------------------------------------------------------------------
__global__ void gather_glove(const float* __restrict__ glove,
                             const int* __restrict__ qf, float* __restrict__ out) {
  int i = blockIdx.x * blockDim.x + threadIdx.x;
  if (i >= B_ * LQ_ * KPAD_) return;
  int r = i / KPAD_, c = i % KPAD_;
  out[i] = (c < 300) ? glove[(size_t)qf[r] * 300 + c] : 0.f;
}

__global__ void pad_sybW(const float* __restrict__ w, float* __restrict__ out) {
  int i = blockIdx.x * blockDim.x + threadIdx.x;
  if (i >= KPAD_ * 2048) return;
  int r = i / 2048, c = i % 2048;
  out[i] = (r < 300) ? w[r * 2048 + c] : 0.f;
}

__global__ void build_masks(const int* __restrict__ qm, const int* __restrict__ vg,
                            const int* __restrict__ qg,
                            float* __restrict__ mA, float* __restrict__ mB) {
  int i = blockIdx.x * blockDim.x + threadIdx.x;
  if (i >= B_ * SS_ * SS_) return;
  int b = i / (SS_ * SS_);
  int rem = i % (SS_ * SS_);
  int s1 = rem / SS_, s2 = rem % SS_;
  float a = 0.f, bb = 1.f;  // cross blocks of `graph` are 1
  if (s1 >= LV_ && s2 >= LV_) {
    int qi = b * LQ_ * LQ_ + (s1 - LV_) * LQ_ + (s2 - LV_);
    a  = (float)qm[qi];
    bb = (float)qg[qi];
  } else if (s1 < LV_ && s2 < LV_) {
    bb = (float)vg[b * LV_ * LV_ + s1 * LV_ + s2];
  }
  mA[i] = a;
  mB[i] = bb;
}

__global__ void add_pos(float* __restrict__ fea, const float* __restrict__ pos) {
  int i = blockIdx.x * blockDim.x + threadIdx.x;
  if (i >= NTOK_ * H_) return;
  int r = i / H_, c = i % H_;
  fea[i] += pos[(r % SS_) * H_ + c];
}

__global__ void dec_init(const float* __restrict__ emb, const float* __restrict__ pos,
                         float* __restrict__ dec) {
  int i = blockIdx.x * blockDim.x + threadIdx.x;
  if (i >= B_ * H_) return;
  int c = i % H_;
  dec[i] = emb[2 * H_ + c] * 27.712812921102035f + pos[c];  // sqrt(768)
}

// ---------------------------------------------------------------------------
// Host orchestration
// ---------------------------------------------------------------------------
extern "C" void kernel_launch(void* const* d_in, const int* in_sizes, int n_in,
                              void* d_out, int out_size, void* d_ws, size_t ws_size,
                              hipStream_t stream) {
  (void)in_sizes; (void)n_in; (void)out_size; (void)ws_size;

  const float* vis_fea   = (const float*)d_in[0];
  const int*   q_fea     = (const int*)d_in[1];
  const int*   q_mask    = (const int*)d_in[3];
  const int*   vis_graph = (const int*)d_in[4];
  const int*   q_graph   = (const int*)d_in[5];
  const float* glove     = (const float*)d_in[6];
  const float* v_W   = (const float*)d_in[7];
  const float* v_b   = (const float*)d_in[8];
  const float* syb_W = (const float*)d_in[9];
  const float* syb_b = (const float*)d_in[10];
  const float* mlp2_W = (const float*)d_in[11];
  const float* mlp2_b = (const float*)d_in[12];
  const float* syb_pos = (const float*)d_in[13];
  const float* dec_emb = (const float*)d_in[14];
  const float* dec_pos = (const float*)d_in[15];
  const float* enc_Wq = (const float*)d_in[16];
  const float* enc_Wk = (const float*)d_in[17];
  const float* enc_Wv = (const float*)d_in[18];
  const float* enc_bq = (const float*)d_in[19];
  const float* enc_bk = (const float*)d_in[20];
  const float* enc_bv = (const float*)d_in[21];
  const float* dec_s_Wv = (const float*)d_in[24];
  const float* dec_s_bv = (const float*)d_in[27];
  const float* dec_c_Wq = (const float*)d_in[28];
  const float* dec_c_Wk = (const float*)d_in[29];
  const float* dec_c_Wv = (const float*)d_in[30];
  const float* dec_c_bq = (const float*)d_in[31];
  const float* dec_c_bk = (const float*)d_in[32];
  const float* dec_c_bv = (const float*)d_in[33];
  const float* enc_f1W = (const float*)d_in[34];
  const float* enc_f1b = (const float*)d_in[35];
  const float* enc_f2W = (const float*)d_in[36];
  const float* enc_f2b = (const float*)d_in[37];
  const float* dec_f1W = (const float*)d_in[38];
  const float* dec_f1b = (const float*)d_in[39];
  const float* dec_f2W = (const float*)d_in[40];
  const float* dec_f2b = (const float*)d_in[41];

  // Workspace layout (floats). region0: concat+qpad early, reused as FFN hid.
  float* ws = (float*)d_ws;
  size_t off = 0;
  const size_t TOKH = (size_t)NTOK_ * H_;
  const size_t TOKHP = TOKH + 8 * H_;                       // +8 zero pad rows
  float* region0 = ws + off; off += (size_t)NTOK_ * FF_;
  float* concat  = region0;                                 // [13824,2048]
  float* qpad    = region0 + (size_t)NTOK_ * 2048;          // [1280,320]
  float* hid     = region0;                                 // [13824,3072] later
  float* fea = ws + off; off += TOKH;
  float* qb_ = ws + off; off += TOKHP;  // Q proj; attention output aliases this
  float* kb_ = ws + off; off += TOKHP;  // K proj; FFN2 output reuses this
  float* vb_ = ws + off; off += TOKHP;  // V proj
  float* mA  = ws + off; off += (size_t)B_ * SS_ * SS_;
  float* mB  = ws + off; off += (size_t)B_ * SS_ * SS_;
  float* dec  = ws + off; off += (size_t)B_ * H_;
  float* dQ   = ws + off; off += (size_t)B_ * H_;
  float* dtmp = ws + off; off += (size_t)B_ * H_;
  float* dh   = ws + off; off += (size_t)B_ * FF_;
  float* sybp = ws + off; off += (size_t)KPAD_ * 2048;

  auto gemm = [&](const float* A, const float* Bw, const float* bias, float* C,
                  int M, int N, int K, int relu, int rpb, int obs, int roff) {
    dim3 g((unsigned)(N / 64), (unsigned)(M / 64));
    gemm_bf16_wmma<<<g, 128, 0, stream>>>(A, Bw, bias, C, M, N, K, relu, rpb, obs, roff);
  };

  // Zero the 8 pad rows of Q/K/V buffers (read past row 13823 by last batch)
  hipMemsetAsync(qb_ + TOKH, 0, 8 * H_ * sizeof(float), stream);
  hipMemsetAsync(kb_ + TOKH, 0, 8 * H_ * sizeof(float), stream);
  hipMemsetAsync(vb_ + TOKH, 0, 8 * H_ * sizeof(float), stream);

  // ---- masks + front-end ----
  build_masks<<<(B_ * SS_ * SS_ + 255) / 256, 256, 0, stream>>>(q_mask, vis_graph, q_graph, mA, mB);
  gather_glove<<<(B_ * LQ_ * KPAD_ + 255) / 256, 256, 0, stream>>>(glove, q_fea, qpad);
  pad_sybW<<<(KPAD_ * 2048 + 255) / 256, 256, 0, stream>>>(syb_W, sybp);
  gemm(vis_fea, v_W, v_b, concat, B_ * LV_, 2048, 2048, 1, LV_, SS_, 0);
  gemm(qpad, sybp, syb_b, concat, B_ * LQ_, 2048, KPAD_, 1, LQ_, SS_, LV_);
  gemm(concat, mlp2_W, mlp2_b, fea, NTOK_, H_, 2048, 0, NTOK_, 0, 0);
  add_pos<<<(NTOK_ * H_ + 255) / 256, 256, 0, stream>>>(fea, syb_pos);

  const int lnGrid = (NTOK_ + 7) / 8;

  // ---- encoder ----
  for (int i = 0; i < NB_; ++i) {
    const size_t wo = (size_t)i * H_ * H_;
    gemm(fea, enc_Wq + wo, enc_bq + i * H_, qb_, NTOK_, H_, H_, 1, NTOK_, 0, 0);
    gemm(fea, enc_Wk + wo, enc_bk + i * H_, kb_, NTOK_, H_, H_, 1, NTOK_, 0, 0);
    gemm(fea, enc_Wv + wo, enc_bv + i * H_, vb_, NTOK_, H_, H_, 1, NTOK_, 0, 0);
    attn_wmma<<<B_ * NH_ * QB_, 32, 0, stream>>>(qb_, kb_, vb_, (i < 2 ? mA : mB), qb_);
    ln_residual<<<lnGrid, 256, 0, stream>>>(fea, qb_, fea, NTOK_);
    gemm(fea, enc_f1W + (size_t)i * H_ * FF_, enc_f1b + i * FF_, hid, NTOK_, FF_, H_, 1, NTOK_, 0, 0);
    gemm(hid, enc_f2W + (size_t)i * FF_ * H_, enc_f2b + i * H_, kb_, NTOK_, H_, FF_, 0, NTOK_, 0, 0);
    ln_residual<<<lnGrid, 256, 0, stream>>>(fea, kb_, fea, NTOK_);
  }

  // ---- decoder (1 BOS token per sample) ----
  dec_init<<<(B_ * H_ + 255) / 256, 256, 0, stream>>>(dec_emb, dec_pos, dec);
  for (int i = 0; i < NB_; ++i) {
    const size_t wo = (size_t)i * H_ * H_;
    // self-attention over L=1: softmax == 1 -> o = relu(dec@Wv + bv)
    gemm(dec, dec_s_Wv + wo, dec_s_bv + i * H_, dtmp, B_, H_, H_, 1, B_, 0, 0);
    ln_residual<<<8, 256, 0, stream>>>(dec, dtmp, dec, B_);
    // cross attention vs encoder features
    gemm(dec, dec_c_Wq + wo, dec_c_bq + i * H_, dQ, B_, H_, H_, 1, B_, 0, 0);
    gemm(fea, dec_c_Wk + wo, dec_c_bk + i * H_, kb_, NTOK_, H_, H_, 1, NTOK_, 0, 0);
    gemm(fea, dec_c_Wv + wo, dec_c_bv + i * H_, vb_, NTOK_, H_, H_, 1, NTOK_, 0, 0);
    dec_cross_attn<<<B_ * NH_, 32, 0, stream>>>(dQ, kb_, vb_, dtmp);
    ln_residual<<<8, 256, 0, stream>>>(dec, dtmp, dec, B_);
    // FFN
    gemm(dec, dec_f1W + (size_t)i * H_ * FF_, dec_f1b + i * FF_, dh, B_, FF_, H_, 1, B_, 0, 0);
    gemm(dh, dec_f2W + (size_t)i * FF_ * H_, dec_f2b + i * H_, dtmp, B_, H_, FF_, 0, B_, 0, 0);
    ln_residual<<<8, 256, 0, stream>>>(dec, dtmp, dec, B_);
  }

  hipMemcpyAsync(d_out, dec, (size_t)B_ * H_ * sizeof(float),
                 hipMemcpyDeviceToDevice, stream);
}
